// TestRNN_21964462752069
// MI455X (gfx1250) — matrix operand order repaired and checked by reading, not compiled, not run on hardware
//
#include <hip/hip_runtime.h>
#include <hip/hip_bf16.h>

typedef __attribute__((ext_vector_type(16))) _Float16 v16h;
typedef __attribute__((ext_vector_type(8)))  _Float16 v8h;
typedef __attribute__((ext_vector_type(8)))  float    v8f;

#define NSEQ 2048            // B*R*C sequences
#define TSTEPS 32            // tokens per cell
#define EDIM 256             // embedding dim
#define HDIM 512             // hidden dim
#define NTROWS (NSEQ * TSTEPS)

// ---------------------------------------------------------------------------
// WMMA fragment loaders (CDNA5 v_wmma_f32_16x16x32_f16, wave32)
// A (16x32, f16): lane holds row M=lane&15; K = {8*half+0..7} ∪ {16+8*half+0..7}
// B (32x16, f16): lane holds col N=lane&15; K = {8*half+0..7} ∪ {16+8*half+0..7}
// C/D (16x16 f32): vgpr v -> row M = v + 8*half, col N = lane&15
// ---------------------------------------------------------------------------
__device__ __forceinline__ v16h load_a_frag(const _Float16* __restrict__ A,
                                            int lda, int m0, int k0, int lane) {
  const int half = lane >> 4, row = lane & 15;
  const _Float16* ap = A + (size_t)(m0 + row) * lda + k0;
  v8h lo = *(const v8h*)(ap + 8 * half);
  v8h hi = *(const v8h*)(ap + 16 + 8 * half);
  v16h a;
#pragma unroll
  for (int i = 0; i < 8; ++i) { a[i] = lo[i]; a[i + 8] = hi[i]; }
  return a;
}

__device__ __forceinline__ v16h load_b_frag(const _Float16* __restrict__ W,
                                            int ldw, int n0, int k0, int lane) {
  const int half = lane >> 4, col = lane & 15;
  const _Float16* wp = W + (size_t)(n0 + col) * ldw + k0;
  v8h lo = *(const v8h*)(wp + 8 * half);
  v8h hi = *(const v8h*)(wp + 16 + 8 * half);
  v16h b;
#pragma unroll
  for (int i = 0; i < 8; ++i) { b[i] = lo[i]; b[i + 8] = hi[i]; }
  return b;
}

// ---------------------------------------------------------------------------
// Input projection GEMM:  out_f16[m][n] = sum_k A[m][k]*W[n][k] + ba[n] + bb[n]
// Block: 256 thr = 8 waves as 4(m) x 2(n); wave tile 64x32; block tile 256x64.
// ---------------------------------------------------------------------------
__global__ void __launch_bounds__(256)
gemm_proj_kernel(const _Float16* __restrict__ A, const _Float16* __restrict__ W,
                 const float* __restrict__ bias_a, const float* __restrict__ bias_b,
                 _Float16* __restrict__ out, int K) {
  const int lane = threadIdx.x & 31;
  const int wave = threadIdx.x >> 5;
  const int wm = wave & 3, wn = wave >> 2;
  const int m_base = blockIdx.x * 256 + wm * 64;
  const int n_base = blockIdx.y * 64 + wn * 32;

  v8f acc[4][2];
#pragma unroll
  for (int mt = 0; mt < 4; ++mt)
#pragma unroll
    for (int nt = 0; nt < 2; ++nt) acc[mt][nt] = (v8f)0.0f;

  for (int k0 = 0; k0 < K; k0 += 32) {
    v16h b0 = load_b_frag(W, K, n_base, k0, lane);
    v16h b1 = load_b_frag(W, K, n_base + 16, k0, lane);
#pragma unroll
    for (int mt = 0; mt < 4; ++mt) {
      v16h a = load_a_frag(A, K, m_base + mt * 16, k0, lane);
      acc[mt][0] = __builtin_amdgcn_wmma_f32_16x16x32_f16(
          false, a, false, b0, (short)0, acc[mt][0], false, false);
      acc[mt][1] = __builtin_amdgcn_wmma_f32_16x16x32_f16(
          false, a, false, b1, (short)0, acc[mt][1], false, false);
    }
  }

  const int half = lane >> 4, col = lane & 15;
#pragma unroll
  for (int nt = 0; nt < 2; ++nt) {
    const int n = n_base + nt * 16 + col;
    const float bv = bias_a[n] + bias_b[n];
#pragma unroll
    for (int mt = 0; mt < 4; ++mt) {
      const int mrow = m_base + mt * 16 + half * 8;
      _Float16* op = out + (size_t)mrow * HDIM + n;
#pragma unroll
      for (int v = 0; v < 8; ++v)
        op[(size_t)v * HDIM] = (_Float16)(acc[mt][nt][v] + bv);
    }
  }
}

// ---------------------------------------------------------------------------
// Recurrent step:  PreH[m][n] = relu(PreH[m][n] + sum_k Hprev[m][k]*W[n][k])
// In-place over the pre-activation slab; result is f16 hidden state.
// M = NSEQ (2048), K = N = HDIM (512).  first==1 -> h_prev = 0 (skip matmul).
// ---------------------------------------------------------------------------
__global__ void __launch_bounds__(256)
rnn_step_kernel(const _Float16* __restrict__ Hprev, const _Float16* __restrict__ W,
                _Float16* __restrict__ PreH, int first) {
  const int lane = threadIdx.x & 31;
  const int wave = threadIdx.x >> 5;
  const int wm = wave & 3, wn = wave >> 2;
  const int m_base = blockIdx.x * 256 + wm * 64;
  const int n_base = blockIdx.y * 64 + wn * 32;

  v8f acc[4][2];
#pragma unroll
  for (int mt = 0; mt < 4; ++mt)
#pragma unroll
    for (int nt = 0; nt < 2; ++nt) acc[mt][nt] = (v8f)0.0f;

  if (!first) {
    for (int k0 = 0; k0 < HDIM; k0 += 32) {
      v16h b0 = load_b_frag(W, HDIM, n_base, k0, lane);
      v16h b1 = load_b_frag(W, HDIM, n_base + 16, k0, lane);
#pragma unroll
      for (int mt = 0; mt < 4; ++mt) {
        v16h a = load_a_frag(Hprev, HDIM, m_base + mt * 16, k0, lane);
        acc[mt][0] = __builtin_amdgcn_wmma_f32_16x16x32_f16(
            false, a, false, b0, (short)0, acc[mt][0], false, false);
        acc[mt][1] = __builtin_amdgcn_wmma_f32_16x16x32_f16(
            false, a, false, b1, (short)0, acc[mt][1], false, false);
      }
    }
  }

  const int half = lane >> 4, col = lane & 15;
#pragma unroll
  for (int nt = 0; nt < 2; ++nt) {
    const int n = n_base + nt * 16 + col;
#pragma unroll
    for (int mt = 0; mt < 4; ++mt) {
      const int mrow = m_base + mt * 16 + half * 8;
      _Float16* op = PreH + (size_t)mrow * HDIM + n;
#pragma unroll
      for (int v = 0; v < 8; ++v) {
        float x = acc[mt][nt][v] + (float)op[(size_t)v * HDIM];
        x = x > 0.0f ? x : 0.0f;
        op[(size_t)v * HDIM] = (_Float16)x;
      }
    }
  }
}

// ---------------------------------------------------------------------------
// Embedding gather -> f16 activation matrix, time-major: row m = t*NSEQ + n
// ---------------------------------------------------------------------------
__global__ void __launch_bounds__(256)
gather_embed_kernel(const int* __restrict__ tok, const float* __restrict__ embed,
                    _Float16* __restrict__ Aemb) {
  const int m = blockIdx.x;          // t*NSEQ + n
  const int t = m / NSEQ, n = m % NSEQ;
  const int v = tok[(size_t)n * TSTEPS + t];
  const int e = threadIdx.x;         // 256 threads, EDIM elems
  Aemb[(size_t)m * EDIM + e] = (_Float16)embed[(size_t)v * EDIM + e];
}

// ---------------------------------------------------------------------------
// fp32 -> f16 weight conversion
// ---------------------------------------------------------------------------
__global__ void __launch_bounds__(256)
f32_to_f16_kernel(const float* __restrict__ src, _Float16* __restrict__ dst, int n) {
  int i = blockIdx.x * 256 + threadIdx.x;
  if (i < n) dst[i] = (_Float16)src[i];
}

// ---------------------------------------------------------------------------
// Head: s[b,c] = dot(w[0:512], h_c) + dot(w[512:1024], S_b - h_c) + b_pred
// One block per batch (8 blocks x 256 threads, one cell per thread).
// ---------------------------------------------------------------------------
__global__ void __launch_bounds__(256)
head_kernel(const _Float16* __restrict__ Hfin, const float* __restrict__ Wp,
            const float* __restrict__ bp, float* __restrict__ out) {
  __shared__ float sw[2 * HDIM];
  __shared__ float sred[256];
  const int tid = threadIdx.x;
  for (int i = tid; i < 2 * HDIM; i += 256) sw[i] = Wp[i];
  __syncthreads();

  const int cell = blockIdx.x * 256 + tid;
  const _Float16* h = Hfin + (size_t)cell * HDIM;
  float d0 = 0.0f, d1 = 0.0f;
  for (int k = 0; k < HDIM; ++k) {
    float hv = (float)h[k];
    d0 += sw[k] * hv;
    d1 += sw[HDIM + k] * hv;
  }
  sred[tid] = d1;
  __syncthreads();
  for (int s = 128; s > 0; s >>= 1) {
    if (tid < s) sred[tid] += sred[tid + s];
    __syncthreads();
  }
  const float S1 = sred[0];
  out[cell] = d0 + (S1 - d1) + bp[0];
}

// ---------------------------------------------------------------------------
extern "C" void kernel_launch(void* const* d_in, const int* in_sizes, int n_in,
                              void* d_out, int out_size, void* d_ws, size_t ws_size,
                              hipStream_t stream) {
  (void)in_sizes; (void)n_in; (void)out_size; (void)ws_size;

  const int*   x     = (const int*)d_in[0];
  const float* embed = (const float*)d_in[1];
  const float* Wih0  = (const float*)d_in[2];
  const float* Whh0  = (const float*)d_in[3];
  const float* bih0  = (const float*)d_in[4];
  const float* bhh0  = (const float*)d_in[5];
  const float* Wih1  = (const float*)d_in[6];
  const float* Whh1  = (const float*)d_in[7];
  const float* bih1  = (const float*)d_in[8];
  const float* bhh1  = (const float*)d_in[9];
  const float* Wp    = (const float*)d_in[10];
  const float* bp    = (const float*)d_in[11];
  float* out = (float*)d_out;

  char* wsp = (char*)d_ws;
  size_t off = 0;
  auto alloc = [&](size_t bytes) -> void* {
    void* p = wsp + off;
    off += (bytes + (size_t)255) & ~(size_t)255;
    return p;
  };

  _Float16* Wih0h = (_Float16*)alloc((size_t)HDIM * EDIM * 2);
  _Float16* Whh0h = (_Float16*)alloc((size_t)HDIM * HDIM * 2);
  _Float16* Wih1h = (_Float16*)alloc((size_t)HDIM * HDIM * 2);
  _Float16* Whh1h = (_Float16*)alloc((size_t)HDIM * HDIM * 2);
  _Float16* Aemb  = (_Float16*)alloc((size_t)NTROWS * EDIM * 2);
  _Float16* P0    = (_Float16*)alloc((size_t)NTROWS * HDIM * 2);  // layer-0 pre/out slab
  _Float16* P1    = (_Float16*)alloc((size_t)NTROWS * HDIM * 2);  // layer-1 pre/out slab

  // 1) weights -> f16 (layouts already [n][k] row-major == WMMA B layout)
  f32_to_f16_kernel<<<(HDIM * EDIM + 255) / 256, 256, 0, stream>>>(Wih0, Wih0h, HDIM * EDIM);
  f32_to_f16_kernel<<<(HDIM * HDIM + 255) / 256, 256, 0, stream>>>(Whh0, Whh0h, HDIM * HDIM);
  f32_to_f16_kernel<<<(HDIM * HDIM + 255) / 256, 256, 0, stream>>>(Wih1, Wih1h, HDIM * HDIM);
  f32_to_f16_kernel<<<(HDIM * HDIM + 255) / 256, 256, 0, stream>>>(Whh1, Whh1h, HDIM * HDIM);

  // 2) embedding gather (time-major rows)
  gather_embed_kernel<<<NTROWS, 256, 0, stream>>>(x, embed, Aemb);

  // 3) layer-0 input projection (one big WMMA GEMM), biases folded in
  gemm_proj_kernel<<<dim3(NTROWS / 256, HDIM / 64), 256, 0, stream>>>(
      Aemb, Wih0h, bih0, bhh0, P0, EDIM);

  // 4) layer-0 recurrence (sequential over T, in-place on P0)
  for (int t = 0; t < TSTEPS; ++t) {
    const _Float16* hprev = (t == 0) ? P0 : (P0 + (size_t)(t - 1) * NSEQ * HDIM);
    rnn_step_kernel<<<dim3(NSEQ / 256, HDIM / 64), 256, 0, stream>>>(
        hprev, Whh0h, P0 + (size_t)t * NSEQ * HDIM, t == 0 ? 1 : 0);
  }

  // 5) layer-1 input projection over all T*N rows of layer-0 output
  gemm_proj_kernel<<<dim3(NTROWS / 256, HDIM / 64), 256, 0, stream>>>(
      P0, Wih1h, bih1, bhh1, P1, HDIM);

  // 6) layer-1 recurrence
  for (int t = 0; t < TSTEPS; ++t) {
    const _Float16* hprev = (t == 0) ? P1 : (P1 + (size_t)(t - 1) * NSEQ * HDIM);
    rnn_step_kernel<<<dim3(NSEQ / 256, HDIM / 64), 256, 0, stream>>>(
        hprev, Whh1h, P1 + (size_t)t * NSEQ * HDIM, t == 0 ? 1 : 0);
  }

  // 7) head: leave-one-out pooled linear predictor
  head_kernel<<<8, 256, 0, stream>>>(
      P1 + (size_t)(TSTEPS - 1) * NSEQ * HDIM, Wp, bp, out);
}
